// WindowAttention_46566035423551
// MI455X (gfx1250) — compile-verified, matrix operand order checked by reading, not run on hardware
//
#include <hip/hip_runtime.h>
#include <hip/hip_bf16.h>
#include <cstdint>

typedef __attribute__((ext_vector_type(16))) _Float16 v16h;
typedef __attribute__((ext_vector_type(8)))  _Float16 v8h;
typedef __attribute__((ext_vector_type(8)))  float    v8f;
typedef __attribute__((ext_vector_type(4)))  int      v4i;

#define N_TOK   64
#define DIM     192
#define HEADS   6
#define HD      32
#define NW      1024
#define LDP     200          // padded row stride (elements) for 64x192 tiles
#define LDV     72           // padded token stride for transposed V [192][72]

// ---- dynamic LDS layout (bytes) ----
// region A: [0,102400)        stage1: lQ f32[64][200] @0, lK f32[64][200] @51200
//                             stage3+: S f32[6][64][64] (98304 B)
// region V: [102400,130048)   lVt f16[192][72]  (V transposed: [channel][token])
// region C: [130048,155648)   lX f16 (stage0/1) -> lQn f16 (stage2/3) -> O f16 (stage5/6)
// region D: [155648,181248)   lKn f16[64][200]
// region P: [181248,230400)   lMask f32[64][64] (stage0-3, async-staged) -> P f16[6][64][64]
#define SMEM_BYTES 230400

#if defined(__has_builtin)
#if __has_builtin(__builtin_amdgcn_global_load_async_to_lds_b128) && \
    __has_builtin(__builtin_amdgcn_s_wait_asynccnt)
#define HAVE_ASYNC_LDS 1
#endif
#endif

#if HAVE_ASYNC_LDS
typedef __attribute__((address_space(1))) v4i* gv4i_p;   // global int4*
typedef __attribute__((address_space(3))) v4i* lv4i_p;   // LDS int4*
#endif

// ---------------- WMMA fragment helpers (ISA 7.12.2 f16 layouts) ----------------
__device__ __forceinline__ int frag_k(int e, int lane) {
  int r = e >> 1;                                   // 32-bit reg index 0..7
  return ((r & 3) << 1) + ((lane >> 4) << 3) + ((r >> 2) << 4) + (e & 1);
}

// A fragment 16x32: A[m][k], m = row0 + lane%16 (two contiguous 8-half runs -> ds_load_b128 x2)
__device__ __forceinline__ v16h frag_a_lds(const _Float16* s, int ld, int row0, int col0, int lane) {
  int m = row0 + (lane & 15);
  v16h a;
#pragma unroll
  for (int e = 0; e < 16; ++e) a[e] = s[m * ld + col0 + frag_k(e, lane)];
  return a;
}

// B fragment 32x16 from N-major rows (implicit transpose): B[k][n] = s[(n0+n)*ld + k0+k]
__device__ __forceinline__ v16h frag_b_nrow_lds(const _Float16* s, int ld, int k0, int n0, int lane) {
  int n = n0 + (lane & 15);
  v16h b;
#pragma unroll
  for (int e = 0; e < 16; ++e) b[e] = s[n * ld + k0 + frag_k(e, lane)];
  return b;
}

// B fragment from fragment-packed global weights: one contiguous 32B load per lane
// packed layout: [kk][nt][lane][16 halves]
__device__ __forceinline__ v16h frag_b_packed(const _Float16* g, int kk, int nt, int lane) {
  return *(const v16h*)(g + (size_t)(((kk * 12 + nt) * 32 + lane) << 4));
}

__device__ __forceinline__ v8f wmma_f16(v16h a, v16h b, v8f c) {
  return __builtin_amdgcn_wmma_f32_16x16x32_f16(false, a, false, b, (short)0, c, false, false);
}

// ---------------- prep: fp32 weights -> f16 fragment-packed (qw|kw|vw|pw) ----------------
// dst[(((m*6+kk)*12+nt)*32+lane)*16+e] = W_m[kk*32+frag_k(e,lane)][nt*16+(lane&15)]
__global__ void pack_w_kernel(const float* __restrict__ qw, const float* __restrict__ kw,
                              const float* __restrict__ vw, const float* __restrict__ pw,
                              _Float16* __restrict__ dst) {
  int i = blockIdx.x * blockDim.x + threadIdx.x;
  if (i >= 4 * DIM * DIM) return;
  int e    = i & 15;
  int lane = (i >> 4) & 31;
  int nt   = (i >> 9) % 12;
  int kk   = (i / (512 * 12)) % 6;
  int m    = i / (DIM * DIM);
  const float* s = (m == 0) ? qw : (m == 1) ? kw : (m == 2) ? vw : pw;
  int k = kk * 32 + frag_k(e, lane);
  int n = nt * 16 + (lane & 15);
  dst[i] = (_Float16)s[k * DIM + n];
}

// ---------------- prep: continuous-position-bias table [225][6] = 16*sigmoid(MLP) ----------------
__device__ __forceinline__ float cpb_coord(float u) {
  float v = u * (8.0f / 7.0f);
  return copysignf(log2f(fabsf(v) + 1.0f) * (1.0f / 3.0f), v);  // sign*log2(|v|+1)/log2(8)
}

__global__ void cpb_kernel(const float* __restrict__ fc1_w, const float* __restrict__ fc1_b,
                           const float* __restrict__ fc2_w, float* __restrict__ bias_tab) {
  __shared__ float acc[HEADS];
  int e  = blockIdx.x;                 // 0..224
  int dy = e / 15 - 7, dx = e % 15 - 7;
  float x0 = cpb_coord((float)dx);     // feature 0 = X (col delta)
  float x1 = cpb_coord((float)dy);     // feature 1 = Y (row delta)
  int j = threadIdx.x;                 // 0..511 hidden units
  if (j < HEADS) acc[j] = 0.0f;
  __syncthreads();
  float h = fmaxf(0.0f, x0 * fc1_w[j] + x1 * fc1_w[512 + j] + fc1_b[j]);
#pragma unroll
  for (int hh = 0; hh < HEADS; ++hh) atomicAdd(&acc[hh], h * fc2_w[j * HEADS + hh]);
  __syncthreads();
  if (j < HEADS) bias_tab[e * HEADS + j] = 16.0f / (1.0f + __expf(-acc[j]));
}

// ---------------- fused window attention: one block per window ----------------
__global__ __launch_bounds__(256, 1)
void swinv2_attn_kernel(const float* __restrict__ x, const float* __restrict__ mask,
                        const float* __restrict__ logit_scale,
                        const float* __restrict__ q_b, const float* __restrict__ v_b,
                        const float* __restrict__ proj_b,
                        const _Float16* __restrict__ wpk,      // fragment-packed qw|kw|vw|pw
                        const float* __restrict__ bias_tab,    // [225][6], 16*sigmoid applied
                        float* __restrict__ out) {
  extern __shared__ char smem[];
  float*    lQ    = (float*)(smem + 0);
  float*    lK    = (float*)(smem + 51200);
  float*    S     = (float*)(smem + 0);
  _Float16* lVt   = (_Float16*)(smem + 102400);   // [192][72] = [channel][token]
  _Float16* lX    = (_Float16*)(smem + 130048);
  _Float16* lQn   = (_Float16*)(smem + 130048);
  _Float16* O     = (_Float16*)(smem + 130048);
  _Float16* lKn   = (_Float16*)(smem + 155648);
  float*    lMask = (float*)(smem + 181248);      // stage0-3; overwritten by P in stage 4
  _Float16* P     = (_Float16*)(smem + 181248);

  const int b    = blockIdx.x;
  const int tid  = threadIdx.x;
  const int wave = tid >> 5;
  const int lane = tid & 31;
  const float* mwin = mask + (size_t)(b & (NW - 1)) * N_TOK * N_TOK;

  // -------- stage 0a: async-stage mask window (16KB raw f32) into LDS; overlaps stages 0-2 --------
#if HAVE_ASYNC_LDS
#pragma unroll
  for (int it = 0; it < 4; ++it) {
    int off = tid * 16 + it * 4096;   // bytes
    __builtin_amdgcn_global_load_async_to_lds_b128(
        (gv4i_p)(uintptr_t)((const char*)mwin + off),
        (lv4i_p)(uint32_t)(uintptr_t)((char*)lMask + off),
        0, 0);
  }
#else
  for (int i = tid; i < N_TOK * N_TOK; i += 256) lMask[i] = mwin[i];
#endif

  // -------- stage 0b: X tile -> LDS (float4 loads, 4-half ds stores); prefetch weights --------
  if (tid < 72) __builtin_prefetch(wpk + (size_t)tid * 2048, 0, 1);  // global_prefetch_b8
  {
    const float4* xb4 = (const float4*)(x + (size_t)b * N_TOK * DIM);
    for (int i = tid; i < N_TOK * DIM / 4; i += 256) {
      float4 v = xb4[i];
      int r = (i * 4) / DIM, c = (i * 4) % DIM;
      _Float16* d = lX + r * LDP + c;
      d[0] = (_Float16)v.x; d[1] = (_Float16)v.y;
      d[2] = (_Float16)v.z; d[3] = (_Float16)v.w;
    }
  }
  __syncthreads();

  // -------- stage 1: Q,K,V = X @ {Wq,Wk,Wv} (+bias). 144 WMMA tiles, 18/wave --------
  for (int t = wave; t < 144; t += 8) {
    int which = t / 48, rem = t % 48, mt = rem / 12, nt = rem % 12;
    const _Float16* W = wpk + which * (DIM * DIM);
    v8f c = {};
#pragma unroll
    for (int kk = 0; kk < 6; ++kk) {
      v16h a  = frag_a_lds(lX, LDP, mt * 16, kk * 32, lane);
      v16h bf = frag_b_packed(W, kk, nt, lane);
      c = wmma_f16(a, bf, c);
    }
    int n     = nt * 16 + (lane & 15);
    int mbase = mt * 16 + ((lane >> 4) << 3);
    if (which == 0) {
      float bias = q_b[n];
#pragma unroll
      for (int j = 0; j < 8; ++j) lQ[(mbase + j) * LDP + n] = c[j] + bias;
    } else if (which == 1) {
#pragma unroll
      for (int j = 0; j < 8; ++j) lK[(mbase + j) * LDP + n] = c[j];
    } else {
      float bias = v_b[n];
      v8h pv;
#pragma unroll
      for (int j = 0; j < 8; ++j) pv[j] = (_Float16)(c[j] + bias);
      *(v8h*)(lVt + (size_t)n * LDV + mbase) = pv;   // transposed, 16B aligned -> ds_store_b128
    }
  }
  __syncthreads();

  // -------- stage 2: cosine-attention normalize q,k per (token, head); fold logit scale into q --------
  for (int rh = tid; rh < N_TOK * HEADS; rh += 256) {
    int token = rh / HEADS, h = rh % HEADS;
    const float* qrow = lQ + token * LDP + h * HD;
    const float* krow = lK + token * LDP + h * HD;
    float sq = 0.0f, sk = 0.0f;
#pragma unroll
    for (int d = 0; d < HD; ++d) { float q = qrow[d]; sq += q * q; float k2 = krow[d]; sk += k2 * k2; }
    float scale = __expf(fminf(logit_scale[h], 4.60517019f));   // min(ls, log(100))
    float rq = rsqrtf(fmaxf(sq, 1e-12f)) * scale;
    float rk = rsqrtf(fmaxf(sk, 1e-12f));
#pragma unroll
    for (int d = 0; d < HD; ++d) {
      lQn[token * LDP + h * HD + d] = (_Float16)(qrow[d] * rq);
      lKn[token * LDP + h * HD + d] = (_Float16)(krow[d] * rk);
    }
  }
#if HAVE_ASYNC_LDS
  __builtin_amdgcn_s_wait_asynccnt(0);   // mask staging complete (per-wave) before barrier
#endif
  __syncthreads();

  // -------- stage 3: S[h] = Qn Kn^T + rel-pos bias + shift mask (from LDS). 96 tiles, 12/wave ----
  for (int t = wave; t < 96; t += 8) {
    int h = t >> 4, rem = t & 15, mt = rem >> 2, nt = rem & 3;
    v16h a  = frag_a_lds(lQn, LDP, mt * 16, h * HD, lane);
    v16h bf = frag_b_nrow_lds(lKn, LDP, h * HD, nt * 16, lane);   // Kn^T via N-major rows
    v8f c = {};
    c = wmma_f16(a, bf, c);
    int kt    = nt * 16 + (lane & 15);
    int mbase = mt * 16 + ((lane >> 4) << 3);
#pragma unroll
    for (int j = 0; j < 8; ++j) {
      int q  = mbase + j;
      int dy = (q >> 3) - (kt >> 3);
      int dx = (q & 7)  - (kt & 7);
      int e  = (dy + 7) * 15 + (dx + 7);                           // REL_POS_INDEX(q,kt)
      S[(h * N_TOK + q) * N_TOK + kt] = c[j] + bias_tab[e * HEADS + h] + lMask[q * N_TOK + kt];
    }
  }
  __syncthreads();

  // -------- stage 4: row softmax (fp32) -> P (f16) --------
  for (int r = tid; r < HEADS * N_TOK; r += 256) {
    const float* row = S + r * N_TOK;
    float mx = -1e30f;
#pragma unroll 4
    for (int k2 = 0; k2 < N_TOK; ++k2) mx = fmaxf(mx, row[k2]);
    float sum = 0.0f;
#pragma unroll 4
    for (int k2 = 0; k2 < N_TOK; ++k2) sum += __expf(row[k2] - mx);
    float inv = 1.0f / sum;
    _Float16* prow = P + r * N_TOK;
#pragma unroll 4
    for (int k2 = 0; k2 < N_TOK; ++k2) prow[k2] = (_Float16)(__expf(row[k2] - mx) * inv);
  }
  __syncthreads();

  // -------- stage 5: O[h] = P[h] @ V[h]. 48 tiles, 6/wave, 2 WMMA each (K=64) --------
  for (int t = wave; t < 48; t += 8) {
    int h = t / 8, rem = t % 8, mt = rem / 2, nt = rem % 2;
    v8f c = {};
#pragma unroll
    for (int kk = 0; kk < 2; ++kk) {
      v16h a  = frag_a_lds(P + h * N_TOK * N_TOK, N_TOK, mt * 16, kk * 32, lane);
      v16h bf = frag_b_nrow_lds(lVt, LDV, kk * 32, h * HD + nt * 16, lane);  // V via transposed store
      c = wmma_f16(a, bf, c);
    }
    int n     = h * HD + nt * 16 + (lane & 15);
    int mbase = mt * 16 + ((lane >> 4) << 3);
#pragma unroll
    for (int j = 0; j < 8; ++j) O[(mbase + j) * LDP + n] = (_Float16)c[j];
  }
  __syncthreads();

  // -------- stage 6: Y = O @ Wproj + b -> global fp32. 48 tiles, 6/wave --------
  const _Float16* pwp = wpk + 3 * (DIM * DIM);
  for (int t = wave; t < 48; t += 8) {
    int mt = t / 12, nt = t % 12;
    v8f c = {};
#pragma unroll
    for (int kk = 0; kk < 6; ++kk) {
      v16h a  = frag_a_lds(O, LDP, mt * 16, kk * 32, lane);
      v16h bf = frag_b_packed(pwp, kk, nt, lane);
      c = wmma_f16(a, bf, c);
    }
    int n     = nt * 16 + (lane & 15);
    float pb  = proj_b[n];
    int mbase = mt * 16 + ((lane >> 4) << 3);
#pragma unroll
    for (int j = 0; j < 8; ++j)
      out[((size_t)b * N_TOK + mbase + j) * DIM + n] = c[j] + pb;
  }
}

extern "C" void kernel_launch(void* const* d_in, const int* in_sizes, int n_in,
                              void* d_out, int out_size, void* d_ws, size_t ws_size,
                              hipStream_t stream) {
  const float* x    = (const float*)d_in[0];
  const float* mask = (const float*)d_in[1];
  const float* ls   = (const float*)d_in[2];
  const float* fc1w = (const float*)d_in[3];
  const float* fc1b = (const float*)d_in[4];
  const float* fc2w = (const float*)d_in[5];
  const float* qw   = (const float*)d_in[6];
  const float* qb   = (const float*)d_in[7];
  const float* kw   = (const float*)d_in[8];
  const float* vw   = (const float*)d_in[9];
  const float* vb   = (const float*)d_in[10];
  const float* pw   = (const float*)d_in[11];
  const float* pb   = (const float*)d_in[12];
  float* out = (float*)d_out;

  // workspace: [0, 294912) fragment-packed f16 weights ; [294912, 300312) bias table f32[225][6]
  _Float16* wpk      = (_Float16*)d_ws;
  float*    bias_tab = (float*)((char*)d_ws + 4 * DIM * DIM * sizeof(_Float16));

  pack_w_kernel<<<(4 * DIM * DIM + 255) / 256, 256, 0, stream>>>(qw, kw, vw, pw, wpk);
  cpb_kernel<<<225, 512, 0, stream>>>(fc1w, fc1b, fc2w, bias_tab);
  swinv2_attn_kernel<<<4096, 256, SMEM_BYTES, stream>>>(x, mask, ls, qb, vb, pb, wpk, bias_tab, out);
}
